// GNNModel_69930657513933
// MI455X (gfx1250) — matrix operand order repaired and checked by reading, not compiled, not run on hardware
//
#include <hip/hip_runtime.h>
#include <cstdint>

#define N_NODES   20000
#define N_EDGES   200000
#define EF_EDGES  220000   // edges + self loops
#define N_GRAPHS  64
#define HID       128
#define HEADS     4
#define HD        512      // HEADS*HID
#define NEG_SLOPE 0.2f

typedef unsigned short u16;
typedef unsigned int   u32;

typedef __attribute__((ext_vector_type(8)))  __bf16 v8bf;
typedef __attribute__((ext_vector_type(16))) __bf16 v16bf;
typedef __attribute__((ext_vector_type(8)))  float  v8f;

union Frag { v16bf v; v8bf h[2]; };

static __device__ __forceinline__ u16 f2bf(float f) {
  u32 u = __float_as_uint(f);
  u32 r = (u + 0x7FFFu + ((u >> 16) & 1u)) >> 16;   // round-to-nearest-even
  return (u16)r;
}
static __device__ __forceinline__ float bf2f(u16 b) {
  return __uint_as_float(((u32)b) << 16);
}
// monotonic float<->uint encoding for atomicMax on floats (handles negatives)
static __device__ __forceinline__ u32 fenc(float f) {
  u32 u = __float_as_uint(f);
  return (u & 0x80000000u) ? ~u : (u | 0x80000000u);
}
static __device__ __forceinline__ float fdec(u32 u) {
  u32 v = (u & 0x80000000u) ? (u & 0x7FFFFFFFu) : ~u;
  return __uint_as_float(v);
}

// ---------------- utility kernels ----------------

__global__ void fill_u32_k(u32* __restrict__ p, u32 val, int n) {
  int i = blockIdx.x * blockDim.x + threadIdx.x;
  if (i < n) p[i] = val;
}

// bf16 transpose of a f32 weight [K, Nout] -> Wt [Nout, K]
__global__ void wtrans_k(const float* __restrict__ W, u16* __restrict__ Wt, int K, int Nout) {
  int i = blockIdx.x * blockDim.x + threadIdx.x;
  if (i >= K * Nout) return;
  int k = i / Nout, n = i - k * Nout;
  Wt[(size_t)n * K + k] = f2bf(W[i]);
}

// build full src/dst arrays with self loops appended
__global__ void srcdst_k(const int* __restrict__ edge_index, int* __restrict__ srcF,
                         int* __restrict__ dstF) {
  int i = blockIdx.x * blockDim.x + threadIdx.x;
  if (i >= EF_EDGES) return;
  if (i < N_EDGES) { srcF[i] = edge_index[i]; dstF[i] = edge_index[N_EDGES + i]; }
  else             { srcF[i] = i - N_EDGES;   dstF[i] = i - N_EDGES; }
}

// h = node_emb[x] -> bf16
__global__ void gather_h_k(const float* __restrict__ emb, const int* __restrict__ x,
                           u16* __restrict__ hb) {
  int i = blockIdx.x * blockDim.x + threadIdx.x;
  if (i >= N_NODES * HID) return;
  int n = i >> 7, c = i & 127;
  hb[i] = f2bf(emb[(size_t)x[n] * HID + c]);
}

// partial sums of rel_emb[edge_attr] over edges (for self-loop mean fill)
__global__ void esum_k(const float* __restrict__ rel, const int* __restrict__ attr,
                       float* __restrict__ esum) {
  int c = threadIdx.x;             // 128 threads
  int e0 = blockIdx.x * 256;
  float s = 0.f;
  for (int j = 0; j < 256; ++j) {
    int e = e0 + j;
    if (e < N_EDGES) s += rel[(size_t)attr[e] * HID + c];
  }
  atomicAdd(&esum[c], s);
}

// e_full (bf16): gathered rel_emb rows + mean rows for self loops
__global__ void efill_k(const float* __restrict__ rel, const int* __restrict__ attr,
                        const float* __restrict__ esum, u16* __restrict__ efb) {
  int i = blockIdx.x * blockDim.x + threadIdx.x;
  if (i >= EF_EDGES * HID) return;
  int e = i >> 7, c = i & 127;
  float v = (e < N_EDGES) ? rel[(size_t)attr[e] * HID + c]
                          : esum[c] * (1.f / (float)N_EDGES);
  efb[i] = f2bf(v);
}

// ---------------- WMMA GEMM: C[M,512] = A[M,K]bf16 @ Wt[512,K]bf16 (+bias) ----------------
// one wave computes a 16x64 tile (4 WMMA tiles sharing the A fragment);
// 8 waves/block -> one block covers 16 rows x 512 cols; grid = M/16.
template <bool BF16_OUT, bool BIAS>
__global__ void gemm_bf16_k(const u16* __restrict__ A, const u16* __restrict__ Wt,
                            const float* __restrict__ bias, void* __restrict__ Cout,
                            int M, int K) {
  const int Nout = HD;
  int wave = threadIdx.x >> 5;
  int lane = threadIdx.x & 31;
  int half = lane >> 4;
  int r    = lane & 15;
  int row_base = blockIdx.x * 16;
  int col_base = wave * 64;

  const u16* arow = A + (size_t)(row_base + r) * K;          // A fragment: row M=r
  const u16* brow0 = Wt + (size_t)(col_base + r) * K;        // B tile t: column col_base+t*16+r
  const u16* brow1 = brow0 + (size_t)16 * K;
  const u16* brow2 = brow0 + (size_t)32 * K;
  const u16* brow3 = brow0 + (size_t)48 * K;

  v8f acc0 = {}, acc1 = {}, acc2 = {}, acc3 = {};
  for (int k0 = 0; k0 < K; k0 += 32) {
    Frag a, b0, b1, b2, b3;
    // A 16x32 bf16 layout: half0 -> K 0..15, half1 -> K 16..31 (contiguous)
    a.h[0] = *(const v8bf*)(arow + k0 + half * 16);
    a.h[1] = *(const v8bf*)(arow + k0 + half * 16 + 8);
    // B 32x16 bf16 layout: lanes 0-15 hold K 0..15, lanes 16-31 hold K 16..31 (contiguous)
    int bo = k0 + half * 16;
    b0.h[0] = *(const v8bf*)(brow0 + bo);  b0.h[1] = *(const v8bf*)(brow0 + bo + 8);
    b1.h[0] = *(const v8bf*)(brow1 + bo);  b1.h[1] = *(const v8bf*)(brow1 + bo + 8);
    b2.h[0] = *(const v8bf*)(brow2 + bo);  b2.h[1] = *(const v8bf*)(brow2 + bo + 8);
    b3.h[0] = *(const v8bf*)(brow3 + bo);  b3.h[1] = *(const v8bf*)(brow3 + bo + 8);
    acc0 = __builtin_amdgcn_wmma_f32_16x16x32_bf16(false, a.v, false, b0.v, (short)0, acc0, false, false);
    acc1 = __builtin_amdgcn_wmma_f32_16x16x32_bf16(false, a.v, false, b1.v, (short)0, acc1, false, false);
    acc2 = __builtin_amdgcn_wmma_f32_16x16x32_bf16(false, a.v, false, b2.v, (short)0, acc2, false, false);
    acc3 = __builtin_amdgcn_wmma_f32_16x16x32_bf16(false, a.v, false, b3.v, (short)0, acc3, false, false);
  }
  int row0 = row_base + half * 8;
#pragma unroll
  for (int t = 0; t < 4; ++t) {
    v8f acc = (t == 0) ? acc0 : (t == 1) ? acc1 : (t == 2) ? acc2 : acc3;
    int col = col_base + t * 16 + r;
    float bv = BIAS ? bias[col] : 0.f;
#pragma unroll
    for (int i = 0; i < 8; ++i) {
      float v = acc[i] + bv;
      if (BF16_OUT) ((u16*)Cout)[(size_t)(row0 + i) * Nout + col] = f2bf(v);
      else          ((float*)Cout)[(size_t)(row0 + i) * Nout + col] = v;
    }
  }
}

// ---------------- attention ----------------

// one wave per (edge, head): alpha = sum_c leakyrelu(xl[src]+xr[dst]+ee) * att ; segment max
__global__ void attn_score_k(const float* __restrict__ xl, const float* __restrict__ xr,
                             const u16* __restrict__ eeb, const int* __restrict__ srcF,
                             const int* __restrict__ dstF, const float* __restrict__ att,
                             float* __restrict__ alpha, u32* __restrict__ amaxU) {
  int gw = (int)((blockIdx.x * blockDim.x + threadIdx.x) >> 5);
  int lane = threadIdx.x & 31;
  if (gw >= EF_EDGES * HEADS) return;
  int e = gw >> 2, h = gw & 3;
  int s = srcF[e], d = dstF[e];
  const float* xls = xl + (size_t)s * HD + h * HID;
  const float* xrd = xr + (size_t)d * HD + h * HID;
  const u16*   ee  = eeb + (size_t)e * HD + h * HID;
  const float* at  = att + h * HID;
  float sum = 0.f;
  for (int c = lane; c < HID; c += 32) {
    float m = xls[c] + xrd[c] + bf2f(ee[c]);
    m = (m > 0.f) ? m : m * NEG_SLOPE;
    sum += m * at[c];
  }
#pragma unroll
  for (int off = 16; off; off >>= 1) sum += __shfl_xor(sum, off, 32);
  if (lane == 0) {
    alpha[(size_t)e * HEADS + h] = sum;
    atomicMax(&amaxU[(size_t)d * HEADS + h], fenc(sum));
  }
}

// ex = exp(alpha - max[dst]); denom[dst] += ex
__global__ void attn_exp_k(float* __restrict__ alpha, const u32* __restrict__ amaxU,
                           const int* __restrict__ dstF, float* __restrict__ denom) {
  int i = blockIdx.x * blockDim.x + threadIdx.x;
  if (i >= EF_EDGES * HEADS) return;
  int e = i >> 2, h = i & 3;
  int d = dstF[e];
  float ex = __expf(alpha[i] - fdec(amaxU[(size_t)d * HEADS + h]));
  alpha[i] = ex;
  atomicAdd(&denom[(size_t)d * HEADS + h], ex);
}

// out[dst] += xl[src] * alpha/(denom[dst]+eps) ; one block per edge
__global__ void aggregate_k(const float* __restrict__ xl, const float* __restrict__ alpha,
                            const float* __restrict__ denom, const int* __restrict__ srcF,
                            const int* __restrict__ dstF, float* __restrict__ agg) {
  int e = blockIdx.x;
  int s = srcF[e], d = dstF[e];
  for (int c = threadIdx.x; c < HD; c += blockDim.x) {
    int h = c >> 7;
    float w = alpha[(size_t)e * HEADS + h] /
              (denom[(size_t)d * HEADS + h] + 1e-16f);
    atomicAdd(&agg[(size_t)d * HD + c], xl[(size_t)s * HD + c] * w);
  }
}

// ---------------- post-processing ----------------

// h1 = elu(agg + bias1) -> bf16 (input to layer-2 GEMMs)
__global__ void post1_k(const float* __restrict__ agg, const float* __restrict__ bias1,
                        u16* __restrict__ h1b) {
  int i = blockIdx.x * blockDim.x + threadIdx.x;
  if (i >= N_NODES * HD) return;
  float v = agg[i] + bias1[i & (HD - 1)];
  v = (v > 0.f) ? v : (__expf(v) - 1.f);
  h1b[i] = f2bf(v);
}

// h2 = elu(mean_heads(agg2) + bias2) (f32, [N,HID])
__global__ void headmean_k(const float* __restrict__ agg, const float* __restrict__ bias2,
                           float* __restrict__ h2) {
  int i = blockIdx.x * blockDim.x + threadIdx.x;
  if (i >= N_NODES * HID) return;
  int n = i >> 7, c = i & 127;
  const float* row = agg + (size_t)n * HD;
  float s = 0.25f * (row[c] + row[HID + c] + row[2 * HID + c] + row[3 * HID + c]) + bias2[c];
  s = (s > 0.f) ? s : (__expf(s) - 1.f);
  h2[i] = s;
}

__global__ void pool_k(const float* __restrict__ h2, const int* __restrict__ batch,
                       float* __restrict__ pooled, float* __restrict__ cnt) {
  int i = blockIdx.x * blockDim.x + threadIdx.x;
  if (i >= N_NODES * HID) return;
  int n = i >> 7, c = i & 127;
  int b = batch[n];
  atomicAdd(&pooled[(size_t)b * HID + c], h2[i]);
  if (c == 0) atomicAdd(&cnt[b], 1.f);
}

// pooled/cnt -> relu(@Wc1+bc1) -> @Wc2+bc2 ; one block (64 thr) per graph
__global__ void classifier_k(const float* __restrict__ pooled, const float* __restrict__ cnt,
                             const float* __restrict__ Wc1, const float* __restrict__ bc1,
                             const float* __restrict__ Wc2, const float* __restrict__ bc2,
                             float* __restrict__ out) {
  int g = blockIdx.x, j = threadIdx.x;   // 64 threads
  __shared__ float sp[HID];
  __shared__ float sh[64];
  float inv = 1.f / fmaxf(cnt[g], 1.f);
  for (int c = j; c < HID; c += 64) sp[c] = pooled[(size_t)g * HID + c] * inv;
  __syncthreads();
  float acc = bc1[j];
  for (int c = 0; c < HID; ++c) acc += sp[c] * Wc1[c * 64 + j];
  sh[j] = fmaxf(acc, 0.f) * Wc2[j];
  __syncthreads();
  if (j == 0) {
    float o = bc2[0];
    for (int k = 0; k < 64; ++k) o += sh[k];
    out[g] = o;
  }
}

// ---------------- launch ----------------

static inline int cdiv(long long a, int b) { return (int)((a + b - 1) / b); }

extern "C" void kernel_launch(void* const* d_in, const int* in_sizes, int n_in,
                              void* d_out, int out_size, void* d_ws, size_t ws_size,
                              hipStream_t stream) {
  const int*   x        = (const int*)  d_in[0];
  const int*   eidx     = (const int*)  d_in[1];
  const int*   eattr    = (const int*)  d_in[2];
  const int*   batch    = (const int*)  d_in[3];
  const float* node_emb = (const float*)d_in[4];
  const float* rel_emb  = (const float*)d_in[5];
  const float* Wl1 = (const float*)d_in[6];  const float* bl1 = (const float*)d_in[7];
  const float* Wr1 = (const float*)d_in[8];  const float* br1 = (const float*)d_in[9];
  const float* We1 = (const float*)d_in[10]; const float* att1 = (const float*)d_in[11];
  const float* bias1 = (const float*)d_in[12];
  const float* Wl2 = (const float*)d_in[13]; const float* bl2 = (const float*)d_in[14];
  const float* Wr2 = (const float*)d_in[15]; const float* br2 = (const float*)d_in[16];
  const float* We2 = (const float*)d_in[17]; const float* att2 = (const float*)d_in[18];
  const float* bias2 = (const float*)d_in[19];
  const float* Wc1 = (const float*)d_in[20]; const float* bc1 = (const float*)d_in[21];
  const float* Wc2 = (const float*)d_in[22]; const float* bc2 = (const float*)d_in[23];
  float* out = (float*)d_out;

  // ---- workspace carve-up (256B aligned regions) ----
  size_t off = 0;
  auto alloc = [&](size_t bytes) -> void* {
    void* p = (char*)d_ws + off;
    off = (off + bytes + 255) & ~(size_t)255;
    return p;
  };
  int*   srcF  = (int*)  alloc(sizeof(int) * EF_EDGES);
  int*   dstF  = (int*)  alloc(sizeof(int) * EF_EDGES);
  float* esum  = (float*)alloc(sizeof(float) * HID);
  u16*   hb    = (u16*)  alloc(sizeof(u16) * (size_t)N_NODES * HID);
  u16*   efb   = (u16*)  alloc(sizeof(u16) * (size_t)EF_EDGES * HID);
  u16*   wlt1  = (u16*)  alloc(sizeof(u16) * HD * HID);
  u16*   wrt1  = (u16*)  alloc(sizeof(u16) * HD * HID);
  u16*   wet1  = (u16*)  alloc(sizeof(u16) * HD * HID);
  u16*   wlt2  = (u16*)  alloc(sizeof(u16) * HD * HD);
  u16*   wrt2  = (u16*)  alloc(sizeof(u16) * HD * HD);
  u16*   wet2  = (u16*)  alloc(sizeof(u16) * HD * HID);
  float* xl    = (float*)alloc(sizeof(float) * (size_t)N_NODES * HD);
  float* xr    = (float*)alloc(sizeof(float) * (size_t)N_NODES * HD);
  u16*   eeb   = (u16*)  alloc(sizeof(u16) * (size_t)EF_EDGES * HD);
  float* alpha = (float*)alloc(sizeof(float) * (size_t)EF_EDGES * HEADS);
  u32*   amaxU = (u32*)  alloc(sizeof(u32) * (size_t)N_NODES * HEADS);
  float* denom = (float*)alloc(sizeof(float) * (size_t)N_NODES * HEADS);
  float* agg   = (float*)alloc(sizeof(float) * (size_t)N_NODES * HD);
  u16*   h1b   = (u16*)  alloc(sizeof(u16) * (size_t)N_NODES * HD);
  float* h2    = (float*)alloc(sizeof(float) * (size_t)N_NODES * HID);
  float* pooled= (float*)alloc(sizeof(float) * N_GRAPHS * HID);
  float* cnt   = (float*)alloc(sizeof(float) * N_GRAPHS);
  (void)n_in; (void)in_sizes; (void)out_size; (void)ws_size;

  const int B = 256;
  const u32 NEG_INF_ENC = ~0xFF800000u;  // fenc(-inf)

  // ---- prologue: conversions / gathers ----
  hipMemsetAsync(esum, 0, sizeof(float) * HID, stream);
  hipMemsetAsync(pooled, 0, sizeof(float) * N_GRAPHS * HID, stream);
  hipMemsetAsync(cnt, 0, sizeof(float) * N_GRAPHS, stream);

  wtrans_k<<<cdiv(HID * HD, B), B, 0, stream>>>(Wl1, wlt1, HID, HD);
  wtrans_k<<<cdiv(HID * HD, B), B, 0, stream>>>(Wr1, wrt1, HID, HD);
  wtrans_k<<<cdiv(HID * HD, B), B, 0, stream>>>(We1, wet1, HID, HD);
  wtrans_k<<<cdiv(HD * HD, B), B, 0, stream>>>(Wl2, wlt2, HD, HD);
  wtrans_k<<<cdiv(HD * HD, B), B, 0, stream>>>(Wr2, wrt2, HD, HD);
  wtrans_k<<<cdiv(HID * HD, B), B, 0, stream>>>(We2, wet2, HID, HD);

  srcdst_k<<<cdiv(EF_EDGES, B), B, 0, stream>>>(eidx, srcF, dstF);
  gather_h_k<<<cdiv((long long)N_NODES * HID, B), B, 0, stream>>>(node_emb, x, hb);
  esum_k<<<cdiv(N_EDGES, 256), HID, 0, stream>>>(rel_emb, eattr, esum);
  efill_k<<<cdiv((long long)EF_EDGES * HID, B), B, 0, stream>>>(rel_emb, eattr, esum, efb);

  // ---- layer 1 GEMMs (WMMA bf16 -> f32) ----
  gemm_bf16_k<false, true><<<N_NODES / 16, B, 0, stream>>>(hb, wlt1, bl1, xl, N_NODES, HID);
  gemm_bf16_k<false, true><<<N_NODES / 16, B, 0, stream>>>(hb, wrt1, br1, xr, N_NODES, HID);
  gemm_bf16_k<true, false><<<EF_EDGES / 16, B, 0, stream>>>(efb, wet1, nullptr, eeb, EF_EDGES, HID);

  // ---- layer 1 attention + aggregation ----
  fill_u32_k<<<cdiv(N_NODES * HEADS, B), B, 0, stream>>>(amaxU, NEG_INF_ENC, N_NODES * HEADS);
  hipMemsetAsync(denom, 0, sizeof(float) * N_NODES * HEADS, stream);
  hipMemsetAsync(agg, 0, sizeof(float) * (size_t)N_NODES * HD, stream);
  attn_score_k<<<cdiv((long long)EF_EDGES * HEADS * 32, B), B, 0, stream>>>(
      xl, xr, eeb, srcF, dstF, att1, alpha, amaxU);
  attn_exp_k<<<cdiv(EF_EDGES * HEADS, B), B, 0, stream>>>(alpha, amaxU, dstF, denom);
  aggregate_k<<<EF_EDGES, B, 0, stream>>>(xl, alpha, denom, srcF, dstF, agg);
  post1_k<<<cdiv((long long)N_NODES * HD, B), B, 0, stream>>>(agg, bias1, h1b);

  // ---- layer 2 GEMMs ----
  gemm_bf16_k<false, true><<<N_NODES / 16, B, 0, stream>>>(h1b, wlt2, bl2, xl, N_NODES, HD);
  gemm_bf16_k<false, true><<<N_NODES / 16, B, 0, stream>>>(h1b, wrt2, br2, xr, N_NODES, HD);
  gemm_bf16_k<true, false><<<EF_EDGES / 16, B, 0, stream>>>(efb, wet2, nullptr, eeb, EF_EDGES, HID);

  // ---- layer 2 attention + aggregation ----
  fill_u32_k<<<cdiv(N_NODES * HEADS, B), B, 0, stream>>>(amaxU, NEG_INF_ENC, N_NODES * HEADS);
  hipMemsetAsync(denom, 0, sizeof(float) * N_NODES * HEADS, stream);
  hipMemsetAsync(agg, 0, sizeof(float) * (size_t)N_NODES * HD, stream);
  attn_score_k<<<cdiv((long long)EF_EDGES * HEADS * 32, B), B, 0, stream>>>(
      xl, xr, eeb, srcF, dstF, att2, alpha, amaxU);
  attn_exp_k<<<cdiv(EF_EDGES * HEADS, B), B, 0, stream>>>(alpha, amaxU, dstF, denom);
  aggregate_k<<<EF_EDGES, B, 0, stream>>>(xl, alpha, denom, srcF, dstF, agg);
  headmean_k<<<cdiv((long long)N_NODES * HID, B), B, 0, stream>>>(agg, bias2, h2);

  // ---- pooling + classifier ----
  pool_k<<<cdiv((long long)N_NODES * HID, B), B, 0, stream>>>(h2, batch, pooled, cnt);
  classifier_k<<<N_GRAPHS, 64, 0, stream>>>(pooled, cnt, Wc1, bc1, Wc2, bc2, out);
}
